// CGATConv_64072322122516
// MI455X (gfx1250) — matrix operand order repaired and checked by reading, not compiled, not run on hardware
//
#include <hip/hip_runtime.h>
#include <hip/hip_bf16.h>
#include <stdint.h>

// ---------- types ----------
typedef __bf16 bf16;
typedef bf16  v16bf __attribute__((ext_vector_type(16)));
typedef bf16  v4bf  __attribute__((ext_vector_type(4)));
typedef float v8f   __attribute__((ext_vector_type(8)));
typedef float v4f   __attribute__((ext_vector_type(4)));
typedef unsigned short u16;
typedef u16 u16x8 __attribute__((ext_vector_type(8)));
typedef unsigned int u32x4 __attribute__((ext_vector_type(4)));

union FragU  { u16x8 u[2]; v16bf v; };
union FragU4 { v4bf  q[4]; v16bf v; };

// f32 -> bf16 RNE, integer path (used only in one-time weight prep)
__device__ __forceinline__ u16 f2bf(float f) {
  unsigned u = __float_as_uint(f);
  unsigned r = u + 0x7FFFu + ((u >> 16) & 1u);
  return (u16)(r >> 16);
}

__device__ __forceinline__ void atomicMaxF(float* addr, float val) {
  if (val >= 0.f) atomicMax((int*)addr, __float_as_int(val));
  else            atomicMin((unsigned int*)addr, (unsigned int)__float_as_int(val));
}

// ---------- async global->LDS copy (CDNA5 ASYNCcnt path), guarded ----------
#if __has_builtin(__builtin_amdgcn_global_load_async_to_lds_b128) && \
    __has_builtin(__builtin_amdgcn_s_wait_asynccnt)
#define USE_ASYNC_CP 1
#else
#define USE_ASYNC_CP 0
#endif

// builtin param type per hipcc diagnostic: pointer to int4 (GCC vector_size)
typedef int v4i __attribute__((vector_size(4 * sizeof(int))));
typedef __attribute__((address_space(1))) v4i* gas_v4i;
typedef __attribute__((address_space(3))) v4i* las_v4i;

// copy 16 bytes global -> LDS
__device__ __forceinline__ void cp16(const void* g, void* l) {
#if USE_ASYNC_CP
  __builtin_amdgcn_global_load_async_to_lds_b128((gas_v4i)(uintptr_t)g, (las_v4i)l,
                                                 0, 0);
#else
  *(u32x4*)l = *(const u32x4*)g;
#endif
}
__device__ __forceinline__ void cp_wait() {
#if USE_ASYNC_CP
  __builtin_amdgcn_s_wait_asynccnt(0);
#endif
}

// ---------- WMMA fragment builders ----------
// A (16x32 bf16, MxK) from an f32 LDS row block; convert via vector fptrunc
// (lowers to v_cvt_pk_bf16_f32, RNE).
// lane L: M=L&15; elems 0..7 -> K=(L>>4)*8+0..7 ; elems 8..15 -> K=16+(L>>4)*8+0..7
__device__ __forceinline__ v16bf a_frag_f32(const float* sArow, int lane, int kt) {
  int row = lane & 15;
  int hi8 = (lane >> 4) * 8;
  const float* p = sArow + row * 128 + kt * 32;
  v4f c0 = *(const v4f*)(p + hi8);
  v4f c1 = *(const v4f*)(p + hi8 + 4);
  v4f c2 = *(const v4f*)(p + 16 + hi8);
  v4f c3 = *(const v4f*)(p + 16 + hi8 + 4);
  FragU4 f;
  f.q[0] = __builtin_convertvector(c0, v4bf);
  f.q[1] = __builtin_convertvector(c1, v4bf);
  f.q[2] = __builtin_convertvector(c2, v4bf);
  f.q[3] = __builtin_convertvector(c3, v4bf);
  return f.v;
}
// B (32x16 bf16, KxN) with B[k][n] = W[n0+n][k], W row-major bf16 (in LDS):
// lane L: N=L&15; elems 0..7 -> K=(L>>4)*8+0..7 ; elems 8..15 -> K=16+(L>>4)*8+0..7
__device__ __forceinline__ v16bf b_frag(const u16* W, int lane, int kt, int nt) {
  int nrow = nt * 16 + (lane & 15);
  int koff = kt * 32 + ((lane >> 4) * 8);
  const u16* p = W + nrow * 128 + koff;
  FragU f;
  f.u[0] = *(const u16x8*)p;
  f.u[1] = *(const u16x8*)(p + 16);
  return f.v;
}

// ---------- kernel 0: init accumulators ----------
__global__ void k_init(float* __restrict__ out1, float* __restrict__ m,
                       float* __restrict__ den, int N) {
  int i = blockIdx.x * 256 + threadIdx.x;
  if (i < N * 16) out1[i] = 0.f;
  if (i < N * 8) { m[i] = __int_as_float(0xFF800000); den[i] = 0.f; }
}

// ---------- kernel 1: fold weights: Wc1=W1@W_ni, Wc2=W2@W_fij, Wc3=W3@W_nj (bf16),
// W_node -> bf16, wsum[d] = sum_h W_attn[h][d] ----------
__global__ void k_prep(const float* __restrict__ W_ni, const float* __restrict__ W_nj,
                       const float* __restrict__ W_fij, const float* __restrict__ W_edges,
                       const float* __restrict__ W_attn, const float* __restrict__ W_node,
                       u16* __restrict__ wc, u16* __restrict__ wnode,
                       float* __restrict__ wsum) {
  int k = threadIdx.x;   // 0..127  (input dim)
  int o = blockIdx.x;    // 0..127  (output dim)
  int m = blockIdx.y;    // 0..3
  if (m < 3) {
    const float* base = (m == 0) ? W_ni : (m == 1 ? W_fij : W_nj);
    float acc = 0.f;
    for (int j = 0; j < 128; ++j)
      acc += W_edges[o * 384 + m * 128 + j] * base[j * 128 + k];
    wc[m * 128 * 128 + o * 128 + k] = f2bf(acc);
    if (m == 0 && o == 0 && k < 16) {
      float s = 0.f;
      for (int h = 0; h < 8; ++h) s += W_attn[h * 16 + k];
      wsum[k] = s;
    }
  } else {
    wnode[o * 128 + k] = f2bf(W_node[o * 128 + k]);
  }
}

// ---------- kernel 2: node GEMMs: t_src / t_dst / h_node (shared A tile) ----------
__global__ void __launch_bounds__(256)
k_node_gemm(const float* __restrict__ nfeats, const u16* __restrict__ wc,
            const u16* __restrict__ wnode, const float* __restrict__ b_node,
            float* __restrict__ t_src, float* __restrict__ t_dst,
            float* __restrict__ h_node, int N) {
  __shared__ float sAf[128 * 128];  // 64KB, f32 tile
  __shared__ u16 sW[128 * 128];     // 32KB, restaged per weight matrix
  int t = threadIdx.x;
  int rowBase = blockIdx.x * 128;
  const float* gA = nfeats + (size_t)rowBase * 128;
  bool full = (rowBase + 128) <= N;
#pragma unroll
  for (int i = 0; i < 16; ++i) {
    int idx = i * 256 + t;                       // 16B chunk id
    if (full || (rowBase + (idx >> 5)) < N) cp16(gA + idx * 4, sAf + idx * 4);
    else { v4f z = {0.f, 0.f, 0.f, 0.f}; *(v4f*)(sAf + idx * 4) = z; }
  }
  if ((size_t)(rowBase + 128) * 128 < (size_t)N * 128)
    __builtin_prefetch(nfeats + (size_t)(rowBase + 128) * 128 + t * 64, 0, 0);

  int wave = t >> 5, lane = t & 31;
  const float* sArow = sAf + (wave * 16) * 128;
  const u16* Wm[3] = { wc, wc + 2 * 128 * 128, wnode }; // Wc1, Wc3, Wnode
  float* outs[3] = { t_src, t_dst, h_node };

  for (int w = 0; w < 3; ++w) {
    // stage weight matrix w into LDS (2048 x 16B)
#pragma unroll
    for (int i = 0; i < 8; ++i) {
      int idx = i * 256 + t;
      cp16(Wm[w] + idx * 8, sW + idx * 8);
    }
    cp_wait();
    __syncthreads();

    v8f acc[8];
    v8f zero = { 0.f, 0.f, 0.f, 0.f, 0.f, 0.f, 0.f, 0.f };
#pragma unroll
    for (int nt = 0; nt < 8; ++nt) acc[nt] = zero;
#pragma unroll
    for (int kt = 0; kt < 4; ++kt) {
      v16bf af = a_frag_f32(sArow, lane, kt);
#pragma unroll
      for (int nt = 0; nt < 8; ++nt) {
        v16bf bfr = b_frag(sW, lane, kt, nt);
        acc[nt] = __builtin_amdgcn_wmma_f32_16x16x32_bf16(
            false, af, false, bfr, (short)0, acc[nt], false, false);
      }
    }
    int half = lane >> 4, nloc = lane & 15;
#pragma unroll
    for (int r = 0; r < 8; ++r) {
      int row = rowBase + wave * 16 + half * 8 + r;
      if (row < N) {
        float* op = outs[w] + (size_t)row * 128;
#pragma unroll
        for (int nt = 0; nt < 8; ++nt) {
          int col = nt * 16 + nloc;
          float v = acc[nt][r];
          if (w == 2) v += b_node[col];
          op[col] = v;
        }
      }
    }
    __syncthreads();   // protect sW before restage
  }
}

// ---------- kernel 3: edge GEMM + gather + leaky_relu + both reductions ----------
__global__ void __launch_bounds__(256)
k_edge_gemm(const float* __restrict__ efeats, const u16* __restrict__ wc2,
            const int* __restrict__ src, const int* __restrict__ dst,
            const float* __restrict__ t_src, const float* __restrict__ t_dst,
            const float* __restrict__ bias, const float* __restrict__ wsum,
            float* __restrict__ aBuf, float* __restrict__ out2, int E) {
  __shared__ float sAf[128 * 128];  // 64KB f32 tile of efeats
  __shared__ u16 sW[128 * 128];     // 32KB Wc2 bf16
  __shared__ int sSrc[128];
  __shared__ int sDst[128];
  int t = threadIdx.x;
  int eBase = blockIdx.x * 128;
  if (t < 128) {
    int e = eBase + t;
    sSrc[t] = (e < E) ? src[e] : 0;
    sDst[t] = (e < E) ? dst[e] : 0;
  }
  // stage Wc2 into LDS
#pragma unroll
  for (int i = 0; i < 8; ++i) {
    int idx = i * 256 + t;
    cp16(wc2 + idx * 8, sW + idx * 8);
  }
  // stage efeats tile (f32) into LDS
  const float* gA = efeats + (size_t)eBase * 128;
  bool full = (eBase + 128) <= E;
#pragma unroll
  for (int i = 0; i < 16; ++i) {
    int idx = i * 256 + t;
    if (full || (eBase + (idx >> 5)) < E) cp16(gA + idx * 4, sAf + idx * 4);
    else { v4f z = {0.f, 0.f, 0.f, 0.f}; *(v4f*)(sAf + idx * 4) = z; }
  }
  // prefetch next block's tile into caches (global_prefetch_b8)
  if (eBase + 256 <= E)
    __builtin_prefetch(efeats + (size_t)(eBase + 128) * 128 + t * 64, 0, 0);
  cp_wait();
  __syncthreads();

  int wave = t >> 5, lane = t & 31;
  const float* sArow = sAf + (wave * 16) * 128;
  v8f acc[8];
  v8f zero = { 0.f, 0.f, 0.f, 0.f, 0.f, 0.f, 0.f, 0.f };
#pragma unroll
  for (int nt = 0; nt < 8; ++nt) acc[nt] = zero;
#pragma unroll
  for (int kt = 0; kt < 4; ++kt) {
    v16bf af = a_frag_f32(sArow, lane, kt);
#pragma unroll
    for (int nt = 0; nt < 8; ++nt) {
      v16bf bfr = b_frag(sW, lane, kt, nt);
      acc[nt] = __builtin_amdgcn_wmma_f32_16x16x32_bf16(
          false, af, false, bfr, (short)0, acc[nt], false, false);
    }
  }

  // epilogue: f_lin = t_e + t_src[src] + t_dst[dst]; leaky_relu; +bias
  // col = nt*16 + (lane&15) => h = nt, d = lane&15
  int half = lane >> 4, nloc = lane & 15;
  float wv = wsum[nloc];
  float bcol[8];
#pragma unroll
  for (int nt = 0; nt < 8; ++nt) bcol[nt] = bias[nt * 16 + nloc];
#pragma unroll
  for (int r = 0; r < 8; ++r) {
    int rloc = wave * 16 + half * 8 + r;
    int e = eBase + rloc;
    int s = sSrc[rloc], d = sDst[rloc];
    const float* ts = t_src + (size_t)s * 128;
    const float* td = t_dst + (size_t)d * 128;
    float o2 = 0.f;
#pragma unroll
    for (int nt = 0; nt < 8; ++nt) {
      int col = nt * 16 + nloc;
      float v = acc[nt][r] + ts[col] + td[col];
      v = (v > 0.f) ? v : 0.01f * v;   // leaky_relu slope 0.01
      v += bcol[nt];
      o2 += v;                          // f_out.sum over heads (h = nt)
      float p = v * wv;                 // a[e,h] partial (sum over d = lanes)
      p += __shfl_xor(p, 1, 32);
      p += __shfl_xor(p, 2, 32);
      p += __shfl_xor(p, 4, 32);
      p += __shfl_xor(p, 8, 32);
      if (nloc == 0 && e < E) aBuf[(size_t)e * 8 + nt] = p;
    }
    if (e < E) out2[(size_t)e * 16 + nloc] = o2;
  }
}

// ---------- kernel 4/5: edge softmax over dst ----------
__global__ void k_attn_max(const float* __restrict__ aBuf, const int* __restrict__ dst,
                           float* __restrict__ m, int E) {
  int i = blockIdx.x * 256 + threadIdx.x;
  if (i < E * 8) {
    int e = i >> 3, h = i & 7;
    atomicMaxF(&m[(size_t)dst[e] * 8 + h], aBuf[i]);
  }
}
__global__ void k_attn_exp(float* __restrict__ aBuf, const int* __restrict__ dst,
                           const float* __restrict__ m, float* __restrict__ den, int E) {
  int i = blockIdx.x * 256 + threadIdx.x;
  if (i < E * 8) {
    int e = i >> 3, h = i & 7;
    float ex = __expf(aBuf[i] - m[(size_t)dst[e] * 8 + h]);
    aBuf[i] = ex;
    atomicAdd(&den[(size_t)dst[e] * 8 + h], ex);
  }
}

// ---------- kernel 6: message = h_node[src]*alpha*norm, segment-sum at dst ----------
__global__ void k_message(const float* __restrict__ exB, const float* __restrict__ den,
                          const float* __restrict__ h_node, const int* __restrict__ src,
                          const int* __restrict__ dst, const float* __restrict__ norm,
                          float* __restrict__ out1, int E) {
  int i = blockIdx.x * 256 + threadIdx.x;
  if (i < E * 16) {
    int e = i >> 4, d = i & 15;
    int s = src[e], dn = dst[e];
    const float* hn = h_node + (size_t)s * 128 + d;
    const float* ex = exB + (size_t)e * 8;
    const float* dd = den + (size_t)dn * 8;
    float acc = 0.f;
#pragma unroll
    for (int h = 0; h < 8; ++h) acc += hn[h * 16] * (ex[h] / dd[h]);
    atomicAdd(&out1[(size_t)dn * 16 + d], acc * norm[e]);
  }
}

// ---------- host ----------
extern "C" void kernel_launch(void* const* d_in, const int* in_sizes, int n_in,
                              void* d_out, int out_size, void* d_ws, size_t ws_size,
                              hipStream_t stream) {
  (void)n_in; (void)out_size; (void)ws_size;
  const float* nfeats  = (const float*)d_in[0];
  const float* efeats  = (const float*)d_in[1];
  const float* norm    = (const float*)d_in[2];
  const int*   src     = (const int*)d_in[3];
  const int*   dst     = (const int*)d_in[4];
  const float* W_ni    = (const float*)d_in[5];
  const float* W_nj    = (const float*)d_in[6];
  const float* W_fij   = (const float*)d_in[7];
  const float* W_edges = (const float*)d_in[8];
  const float* W_attn  = (const float*)d_in[9];
  const float* bias    = (const float*)d_in[10];
  const float* W_node  = (const float*)d_in[11];
  const float* b_node  = (const float*)d_in[12];

  int N = in_sizes[0] / 128;
  int E = in_sizes[2];

  float* out1 = (float*)d_out;
  float* out2 = out1 + (size_t)N * 16;

  // workspace carve (256B aligned slabs)
  char* wsp = (char*)d_ws;
  auto carve = [&](size_t bytes) {
    void* p = (void*)wsp;
    wsp += (bytes + 255) & ~(size_t)255;
    return p;
  };
  float* t_src  = (float*)carve((size_t)N * 128 * 4);
  float* t_dst  = (float*)carve((size_t)N * 128 * 4);
  float* h_node = (float*)carve((size_t)N * 128 * 4);
  float* aBuf   = (float*)carve((size_t)E * 8 * 4);
  float* mBuf   = (float*)carve((size_t)N * 8 * 4);
  float* den    = (float*)carve((size_t)N * 8 * 4);
  u16*   wc     = (u16*)carve((size_t)3 * 128 * 128 * 2);
  u16*   wnode  = (u16*)carve((size_t)128 * 128 * 2);
  float* wsum   = (float*)carve(16 * 4);

  hipLaunchKernelGGL(k_init, dim3((N * 16 + 255) / 256), dim3(256), 0, stream,
                     out1, mBuf, den, N);
  hipLaunchKernelGGL(k_prep, dim3(128, 4), dim3(128), 0, stream,
                     W_ni, W_nj, W_fij, W_edges, W_attn, W_node, wc, wnode, wsum);
  hipLaunchKernelGGL(k_node_gemm, dim3((N + 127) / 128), dim3(256), 0, stream,
                     nfeats, wc, wnode, b_node, t_src, t_dst, h_node, N);
  hipLaunchKernelGGL(k_edge_gemm, dim3((E + 127) / 128), dim3(256), 0, stream,
                     efeats, wc + 128 * 128, src, dst, t_src, t_dst, bias, wsum,
                     aBuf, out2, E);
  hipLaunchKernelGGL(k_attn_max, dim3((E * 8 + 255) / 256), dim3(256), 0, stream,
                     aBuf, dst, mBuf, E);
  hipLaunchKernelGGL(k_attn_exp, dim3((E * 8 + 255) / 256), dim3(256), 0, stream,
                     aBuf, dst, mBuf, den, E);
  hipLaunchKernelGGL(k_message, dim3((E * 16 + 255) / 256), dim3(256), 0, stream,
                     aBuf, den, h_node, src, dst, norm, out1, E);
}
